// GCNExternal_8246337208594
// MI455X (gfx1250) — matrix operand order repaired and checked by reading, not compiled, not run on hardware
//
#include <hip/hip_runtime.h>
#include <cstdint>

typedef __attribute__((ext_vector_type(16))) _Float16 v16h;
typedef __attribute__((ext_vector_type(8)))  float    v8f;

#define HDIM 128

// ---------------- degree / norm kernels ----------------
__global__ void k_deg_init(float* deg, int n) {
  int i = blockIdx.x * blockDim.x + threadIdx.x;
  if (i < n) deg[i] = 1.0f;  // self loop contributes 1
}

__global__ void k_deg_count(const int* __restrict__ dst, float* deg, int e) {
  int i = blockIdx.x * blockDim.x + threadIdx.x;
  if (i < e) atomicAdd(&deg[dst[i]], 1.0f);
}

__global__ void k_dinv(const float* __restrict__ deg, float* dinv, int n) {
  int i = blockIdx.x * blockDim.x + threadIdx.x;
  if (i < n) dinv[i] = rsqrtf(deg[i]);  // deg >= 1 always
}

// acc[i,j] = bias[j] + dinv[i]^2 * xw[i,j]   (bias + self-loop term fused)
__global__ void k_init_acc(const float* __restrict__ xw, const float* __restrict__ bias,
                           const float* __restrict__ dinv, float* __restrict__ out, int n) {
  int i = blockIdx.x * blockDim.x + threadIdx.x;
  if (i < n * HDIM) {
    int row = i >> 7;
    int col = i & 127;
    float d = dinv[row];
    out[i] = bias[col] + d * d * xw[i];
  }
}

// one wave (32 lanes) per edge; float4 gather + 4 fp32 atomics per lane
__global__ __launch_bounds__(256) void k_spmm(const int* __restrict__ src,
                                              const int* __restrict__ dst,
                                              const float* __restrict__ dinv,
                                              const float* __restrict__ x,
                                              float* __restrict__ out, int e) {
  int eid  = blockIdx.x * 8 + (threadIdx.x >> 5);
  int lane = threadIdx.x & 31;
  if (eid >= e) return;
  int s = src[eid];
  int d = dst[eid];
  float w = dinv[s] * dinv[d];
  float4 v = ((const float4*)x)[(size_t)s * 32 + lane];
  float* o = out + (size_t)d * HDIM + lane * 4;
  atomicAdd(o + 0, w * v.x);
  atomicAdd(o + 1, w * v.y);
  atomicAdd(o + 2, w * v.z);
  atomicAdd(o + 3, w * v.w);
}

// ---------------- WMMA GEMM: out[N x 128] = A[N x Ktot] @ W[Ktot x 128] ----------------
// A = [A0 (K0 wide) | A1 (Ktot-K0 wide)] concatenated along K (for the fusion projection).
// reluA: apply ReLU while staging A (folds previous layer's activation).
// epi: 0 = raw store, 1 = +bias, 2 = +bias -> LayerNorm -> ReLU (encoder).
__global__ __launch_bounds__(256) void k_gemm_wmma(
    const float* __restrict__ A0, const float* __restrict__ A1,
    int K0, int Ktot,
    const float* __restrict__ W, const float* __restrict__ bias,
    const float* __restrict__ lng, const float* __restrict__ lnb,
    float* __restrict__ out, int nrows, int reluA, int epi) {
  constexpr int AP = 34;  // padded pitch (halfs): gcd(17,64)=1 -> conflict-free
  constexpr int BP = 34;
  __shared__ _Float16 As[128 * AP];   // row-major [m][k]
  __shared__ _Float16 Bs[128 * BP];   // n-major  [n][k] (paired-K u32 reads)

  const int tid  = threadIdx.x;
  const int wave = tid >> 5;
  const int lane = tid & 31;
  const int hs   = lane >> 4;   // half-wave select
  const int l16  = lane & 15;
  const int rowBase = blockIdx.x * 128;
  const int K1 = Ktot - K0;

  v8f acc[8];
#pragma unroll
  for (int c = 0; c < 8; ++c)
#pragma unroll
    for (int r = 0; r < 8; ++r) acc[c][r] = 0.0f;

  for (int kb = 0; kb < Ktot; kb += 32) {
    // ---- stage A tile (128 x 32), fp32 -> f16, optional ReLU, concat select ----
#pragma unroll
    for (int e2 = 0; e2 < 16; ++e2) {
      int idx = e2 * 256 + tid;          // consecutive tid -> consecutive k (coalesced rows)
      int m = idx >> 5, k = idx & 31;
      int row = rowBase + m;
      int gk  = kb + k;
      float v = 0.0f;
      if (row < nrows)
        v = (gk < K0) ? A0[(size_t)row * K0 + gk]
                      : A1[(size_t)row * K1 + (gk - K0)];
      if (reluA) v = fmaxf(v, 0.0f);
      As[m * AP + k] = (_Float16)v;
    }
    // ---- stage B tile (32 x 128) n-major; consecutive tid -> consecutive n (coalesced) ----
#pragma unroll
    for (int e2 = 0; e2 < 16; ++e2) {
      int idx = e2 * 256 + tid;
      int k = idx >> 7, n = idx & 127;
      Bs[n * BP + k] = (_Float16)W[(size_t)(kb + k) * HDIM + n];
    }
    __syncthreads();

    // ---- A fragment: lanes 0-15 K in {0-7,16-23}; lanes 16-31 K in {8-15,24-31} ----
    v16h a;
    const _Float16* Ap = &As[(wave * 16 + l16) * AP + 8 * hs];
#pragma unroll
    for (int i = 0; i < 8; ++i) {
      int kk = (i < 4) ? 2 * i : 2 * i + 8;
      a[2 * i]     = Ap[kk];
      a[2 * i + 1] = Ap[kk + 1];
    }
    // ---- 8 column tiles: B fragment + WMMA ----
#pragma unroll
    for (int c = 0; c < 8; ++c) {
      v16h b;
      const _Float16* Bp = &Bs[(c * 16 + l16) * BP + 8 * hs];
#pragma unroll
      for (int i = 0; i < 8; ++i) {
        int kk = (i < 4) ? 2 * i : 2 * i + 8;
        b[2 * i]     = Bp[kk];
        b[2 * i + 1] = Bp[kk + 1];
      }
      acc[c] = __builtin_amdgcn_wmma_f32_16x16x32_f16(
          false, a, false, b, (short)0, acc[c], false, false);
    }
    __syncthreads();
  }

  // ---- epilogue. C/D layout: vgpr r -> M = r + 8*hs; N = l16 (per 16-col tile) ----
  const int mrow0 = rowBase + wave * 16 + 8 * hs;
  if (epi == 2) {
    // +bias -> LayerNorm(128) -> ReLU ; row lives in one 16-lane half (8 vals/lane)
#pragma unroll
    for (int r = 0; r < 8; ++r) {
      float v[8], s = 0.0f, ss = 0.0f;
#pragma unroll
      for (int c = 0; c < 8; ++c) {
        v[c] = acc[c][r] + bias[c * 16 + l16];
        s  += v[c];
        ss += v[c] * v[c];
      }
#pragma unroll
      for (int m = 1; m < 16; m <<= 1) {
        s  += __shfl_xor(s,  m, 32);
        ss += __shfl_xor(ss, m, 32);
      }
      float mu  = s * (1.0f / HDIM);
      float var = ss * (1.0f / HDIM) - mu * mu;
      float rs  = rsqrtf(var + 1e-5f);
      int row = mrow0 + r;
      if (row < nrows) {
#pragma unroll
        for (int c = 0; c < 8; ++c) {
          float o = (v[c] - mu) * rs * lng[c * 16 + l16] + lnb[c * 16 + l16];
          out[(size_t)row * HDIM + c * 16 + l16] = fmaxf(o, 0.0f);
        }
      }
    }
  } else if (epi == 1) {
#pragma unroll
    for (int r = 0; r < 8; ++r) {
      int row = mrow0 + r;
      if (row < nrows) {
#pragma unroll
        for (int c = 0; c < 8; ++c)
          out[(size_t)row * HDIM + c * 16 + l16] = acc[c][r] + bias[c * 16 + l16];
      }
    }
  } else {
#pragma unroll
    for (int r = 0; r < 8; ++r) {
      int row = mrow0 + r;
      if (row < nrows) {
#pragma unroll
        for (int c = 0; c < 8; ++c)
          out[(size_t)row * HDIM + c * 16 + l16] = acc[c][r];
      }
    }
  }
}

// ---------------- launch ----------------
extern "C" void kernel_launch(void* const* d_in, const int* in_sizes, int n_in,
                              void* d_out, int out_size, void* d_ws, size_t ws_size,
                              hipStream_t stream) {
  const int*   ei    = (const int*)d_in[0];
  const float* ext   = (const float*)d_in[1];
  const float* emb   = (const float*)d_in[2];
  const float* encW  = (const float*)d_in[3];
  const float* encb  = (const float*)d_in[4];
  const float* lng   = (const float*)d_in[5];
  const float* lnb   = (const float*)d_in[6];
  const float* projW = (const float*)d_in[7];
  const float* projb = (const float*)d_in[8];
  const float* W0 = (const float*)d_in[9];
  const float* b0 = (const float*)d_in[10];
  const float* W1 = (const float*)d_in[11];
  const float* b1 = (const float*)d_in[12];
  const float* W2 = (const float*)d_in[13];
  const float* b2 = (const float*)d_in[14];
  float* out = (float*)d_out;

  const int E    = in_sizes[0] / 2;
  const int N    = in_sizes[2] / HDIM;
  const int EXTD = in_sizes[1] / N;  // 768

  const int* src = ei;
  const int* dst = ei + E;

  float* wsf  = (float*)d_ws;
  float* deg  = wsf;
  float* dinv = wsf + N;
  float* bufA = wsf + 2 * (size_t)N;                 // x after projection / L1 acc
  float* bufB = bufA + (size_t)N * HDIM;             // GEMM (xW) output
  float* bufC = bufB + (size_t)N * HDIM;             // feat / L0 acc

  const int TPB = 256;
  const int gN   = (N + TPB - 1) / TPB;
  const int gE   = (E + TPB - 1) / TPB;
  const int gNH  = (N * HDIM + TPB - 1) / TPB;
  const int gGem = (N + 127) / 128;
  const int gSp  = (E + 7) / 8;

  // 1) GCN edge norms
  k_deg_init <<<gN, TPB, 0, stream>>>(deg, N);
  k_deg_count<<<gE, TPB, 0, stream>>>(dst, deg, E);
  k_dinv     <<<gN, TPB, 0, stream>>>(deg, dinv, N);

  // 2) feat = ReLU(LN(ext @ encW + encb))  -> bufC
  k_gemm_wmma<<<gGem, TPB, 0, stream>>>(ext, ext, EXTD, EXTD, encW, encb,
                                        lng, lnb, bufC, N, 0, 2);

  // 3) x = [emb | feat] @ projW + projb    -> bufA
  k_gemm_wmma<<<gGem, TPB, 0, stream>>>(emb, bufC, HDIM, 2 * HDIM, projW, projb,
                                        lng, lnb, bufA, N, 0, 1);

  // 4) layer 0: xw = x @ W0 -> bufB ; acc = b0 + dinv^2*xw -> bufC ; spmm adds
  k_gemm_wmma<<<gGem, TPB, 0, stream>>>(bufA, bufA, HDIM, HDIM, W0, b0,
                                        lng, lnb, bufB, N, 0, 0);
  k_init_acc <<<gNH, TPB, 0, stream>>>(bufB, b0, dinv, bufC, N);
  k_spmm     <<<gSp, TPB, 0, stream>>>(src, dst, dinv, bufB, bufC, E);

  // 5) layer 1: (ReLU folded into A staging)
  k_gemm_wmma<<<gGem, TPB, 0, stream>>>(bufC, bufC, HDIM, HDIM, W1, b1,
                                        lng, lnb, bufB, N, 1, 0);
  k_init_acc <<<gNH, TPB, 0, stream>>>(bufB, b1, dinv, bufA, N);
  k_spmm     <<<gSp, TPB, 0, stream>>>(src, dst, dinv, bufB, bufA, E);

  // 6) layer 2 (no final ReLU): result straight into d_out
  k_gemm_wmma<<<gGem, TPB, 0, stream>>>(bufA, bufA, HDIM, HDIM, W2, b2,
                                        lng, lnb, bufB, N, 1, 0);
  k_init_acc <<<gNH, TPB, 0, stream>>>(bufB, b2, dinv, out, N);
  k_spmm     <<<gSp, TPB, 0, stream>>>(src, dst, dinv, bufB, out, E);
}
